// NeuralDict_11441792877314
// MI455X (gfx1250) — matrix operand order repaired and checked by reading, not compiled, not run on hardware
//
#include <hip/hip_runtime.h>

typedef __attribute__((ext_vector_type(16))) _Float16 v16h;
typedef __attribute__((ext_vector_type(8)))  float    v8f;

#define EPSN 1e-12f

// ---------------------------------------------------------------------------
// Kernel 1: L2-normalize the query [128 x 128] and pack it (f16) into the
// WMMA B-matrix fragments, in this code's contiguous k order:
//   fragment block (nt, f) = 32 lanes x 16 halfs (32B/lane, contiguous)
//   lane = h*16 + (n & 15), element e -> K = 32*f + 16*h + e, N-tile nt = n>>4
// ---------------------------------------------------------------------------
__global__ __launch_bounds__(128)
void nd_prep_query(const float* __restrict__ query, _Float16* __restrict__ qf)
{
    const int n = threadIdx.x;              // query row 0..127
    const float* qr = query + n * 128;

    float s = 0.f;
    for (int k4 = 0; k4 < 128; k4 += 4) {
        float4 x = *(const float4*)(qr + k4);
        s += x.x * x.x + x.y * x.y + x.z * x.z + x.w * x.w;
    }
    const float scale = 1.0f / fmaxf(sqrtf(s), EPSN);

    const int nt = n >> 4, lid = n & 15;
    for (int k = 0; k < 128; ++k) {
        const int f = k >> 5, r = k & 31, h = r >> 4, pos = r & 15;
        const int lane = h * 16 + lid;
        qf[(nt * 4 + f) * 512 + lane * 16 + pos] = (_Float16)(qr[k] * scale);
    }
}

// ---------------------------------------------------------------------------
// Kernel 2: per block (128 threads = 4 waves) handle 64 key rows.
//  - copy packed query fragments d_ws -> LDS (32KB)
//  - each wave: 16 key rows; A fragments from global f32 (layout-exact),
//    row sumsq on the fly; 32x v_wmma_f32_16x16x32_f16; scale by 1/||key||;
//    park 16x128 f32 cosine tile in per-wave LDS (32KB total)
//  - exact wave-cooperative top-k extraction with index tie-breaking
// ---------------------------------------------------------------------------
__global__ __launch_bounds__(128)
void nd_knn(const float* __restrict__ keys,
            const _Float16* __restrict__ qf,
            float* __restrict__ out_val,
            int*   __restrict__ out_idx,
            int Dn, int k)
{
    __shared__ __align__(32) _Float16 qf_lds[16384];   // 32 KB packed B frags
    __shared__ float cos_lds[4 * 2048];                // 32 KB: 4 waves x 16x128

    const int tid  = threadIdx.x;
    const int wave = tid >> 5;
    const int lane = tid & 31;
    const int h    = lane >> 4;          // half-wave
    const int lid  = lane & 15;

    // ---- stage packed query fragments into LDS (coalesced b128) ----
    {
        const float4* src = (const float4*)qf;
        float4*       dst = (float4*)qf_lds;
        for (int i = tid; i < 2048; i += 128) dst[i] = src[i];
    }
    __syncthreads();

    const int m0 = blockIdx.x * 64 + wave * 16;
    int rowL = m0 + lid; if (rowL > Dn - 1) rowL = Dn - 1;   // clamp for loads
    const float* kr = keys + (long long)rowL * 128;

    // ---- A fragments (contiguous k order, as B): lane (h,lid) loads, per K-frag f,
    //      floats [32f+16h .. +15] of its row ----
    v16h a[4];
    float ss = 0.f;
#pragma unroll
    for (int f = 0; f < 4; ++f) {
        const int base = f * 32 + h * 16;
        float4 x0 = *(const float4*)(kr + base);
        float4 x1 = *(const float4*)(kr + base + 4);
        float4 x2 = *(const float4*)(kr + base + 8);
        float4 x3 = *(const float4*)(kr + base + 12);
        float xs[16] = { x0.x, x0.y, x0.z, x0.w,  x1.x, x1.y, x1.z, x1.w,
                         x2.x, x2.y, x2.z, x2.w,  x3.x, x3.y, x3.z, x3.w };
#pragma unroll
        for (int e = 0; e < 16; ++e) {
            ss += xs[e] * xs[e];
            a[f][e] = (_Float16)xs[e];
        }
    }
    // lanes lid and lid+16 together hold row lid -> combine halves
    ss += __shfl_xor(ss, 16);
    const float scale = 1.0f / fmaxf(sqrtf(ss), EPSN);
    // broadcast: row (v + 8h)'s scale lives on lane (v + 8h)
    float sc[8];
#pragma unroll
    for (int v = 0; v < 8; ++v) sc[v] = __shfl(scale, v + 8 * h);

    // ---- GEMM: 8 N-tiles x 4 K-frags, spill scaled cosine to LDS ----
    float* cw = cos_lds + wave * 2048;
#pragma unroll
    for (int nt = 0; nt < 8; ++nt) {
        v8f acc = {};
#pragma unroll
        for (int f = 0; f < 4; ++f) {
            v16h b = *(const v16h*)(qf_lds + (nt * 4 + f) * 512 + lane * 16);
            acc = __builtin_amdgcn_wmma_f32_16x16x32_f16(
                false, a[f], false, b, (short)0, acc, false, false);
        }
#pragma unroll
        for (int v = 0; v < 8; ++v)                 // C layout: VGPR v -> M=v+8h
            cw[(v + 8 * h) * 128 + nt * 16 + lid] = acc[v] * sc[v];
    }
    // cross-lane LDS handoff within the wave: drain DS before re-reading
    asm volatile("s_wait_dscnt 0" ::: "memory");

    // ---- exact top-k per row (whole wave per row, 4 values/lane) ----
    for (int r = 0; r < 16; ++r) {
        const int row = m0 + r;
        if (row >= Dn) break;                        // wave-uniform
        float v0 = cw[r * 128 + lane];
        float v1 = cw[r * 128 + lane + 32];
        float v2 = cw[r * 128 + lane + 64];
        float v3 = cw[r * 128 + lane + 96];
        const long long ob = (long long)row * k;
        for (int t = 0; t < k; ++t) {
            // local argmax (strict > keeps smallest column on ties)
            float bv = v0; int bi = 0;
            if (v1 > bv) { bv = v1; bi = 1; }
            if (v2 > bv) { bv = v2; bi = 2; }
            if (v3 > bv) { bv = v3; bi = 3; }
            // wave max
            float mx = bv;
#pragma unroll
            for (int s2 = 16; s2 >= 1; s2 >>= 1)
                mx = fmaxf(mx, __shfl_xor(mx, s2));
            // smallest column achieving the max (torch tie-break)
            int mc = (bv == mx) ? (lane + (bi << 5)) : 0x7fffffff;
#pragma unroll
            for (int s2 = 16; s2 >= 1; s2 >>= 1) {
                int o = __shfl_xor(mc, s2);
                mc = (o < mc) ? o : mc;
            }
            // rotating scribe writes (value, index)
            if (lane == (t & 31)) { out_val[ob + t] = mx; out_idx[ob + t] = mc; }
            // winner retires its element
            const bool own = (lane == (mc & 31));
            const int  ii  = mc >> 5;
            v0 = (own && ii == 0) ? -3.0e38f : v0;
            v1 = (own && ii == 1) ? -3.0e38f : v1;
            v2 = (own && ii == 2) ? -3.0e38f : v2;
            v3 = (own && ii == 3) ? -3.0e38f : v3;
        }
    }
}

// ---------------------------------------------------------------------------
extern "C" void kernel_launch(void* const* d_in, const int* in_sizes, int n_in,
                              void* d_out, int out_size, void* d_ws, size_t ws_size,
                              hipStream_t stream)
{
    const float* keys  = (const float*)d_in[0];   // [D, 128] f32
    const float* query = (const float*)d_in[1];   // [128, 128] f32
    (void)n_in; (void)ws_size;

    const int Dn = in_sizes[0] / 128;             // 500000
    const int k  = (Dn > 0) ? (int)(out_size / (2LL * Dn)) : 0;   // 50

    _Float16* qfrag = (_Float16*)d_ws;            // 32 KB packed query frags

    nd_prep_query<<<1, 128, 0, stream>>>(query, qfrag);

    float* out_val = (float*)d_out;                       // [D, k] f32
    int*   out_idx = (int*)d_out + (size_t)Dn * k;        // [D, k] i32 bits

    const int blocks = (Dn + 63) / 64;
    nd_knn<<<blocks, 128, 0, stream>>>(keys, qfrag, out_val, out_idx, Dn, k);
}
